// EncoderBlock_83760452207437
// MI455X (gfx1250) — matrix operand order repaired and checked by reading, not compiled, not run on hardware
//
#include <hip/hip_runtime.h>
#include <hip/hip_bf16.h>

typedef __attribute__((ext_vector_type(16))) _Float16 v16h;
typedef __attribute__((ext_vector_type(8)))  _Float16 v8h;
typedef __attribute__((ext_vector_type(8)))  float    v8f;
typedef __attribute__((ext_vector_type(4)))  unsigned v4u;
typedef __attribute__((ext_vector_type(8)))  int      v8i;
typedef __attribute__((ext_vector_type(4)))  int      v4i;

#define DIV_UP(a,b) (((a)+(b)-1)/(b))

// Tensor Data Mover availability (device pass only; host pass falls back).
#if defined(__has_builtin)
#  if __has_builtin(__builtin_amdgcn_tensor_load_to_lds) && \
      __has_builtin(__builtin_amdgcn_s_wait_tensorcnt)
#    define USE_TDM 1
#  endif
#endif
#ifndef USE_TDM
#  define USE_TDM 0
#endif

#if USE_TDM
// Issue a 2D TDM tile load: tile (tile1 rows x tile0 elems) of a row-major
// [dim1 x dim0] f16 tensor (row stride stride0 elems) -> LDS at lds_off,
// packed contiguously (LDS row stride = tile0*2 bytes).
// D# packing per CDNA5 ISA.
__device__ __forceinline__ void tdm_load_2d_f16(const void* gaddr, unsigned lds_off,
                                                unsigned dim0, unsigned dim1,
                                                unsigned tile0, unsigned tile1,
                                                unsigned long long stride0) {
    unsigned long long ga = (unsigned long long)gaddr;
    v4u g0;
    g0[0] = 1u;                                         // count=1 (valid user D#)
    g0[1] = lds_off;                                    // lds_addr (bytes)
    g0[2] = (unsigned)(ga & 0xffffffffu);               // global_addr[31:0]
    g0[3] = (unsigned)((ga >> 32) & 0x01ffffffu)        // global_addr[56:32]
            | 0x80000000u;                              // type=2 ("image")
    v8i g1;
    g1[0] = 0x00010000;                                 // wg_mask=0, data_size=1 (2B)
    g1[1] = (int)((dim0 & 0xffffu) << 16);              // tensor_dim0[15:0]
    g1[2] = (int)((dim0 >> 16) | ((dim1 & 0xffffu) << 16));
    g1[3] = (int)((dim1 >> 16) | ((tile0 & 0xffffu) << 16));
    g1[4] = (int)(tile1 & 0xffffu);                     // tile_dim1 (tile_dim2=0)
    g1[5] = (int)(stride0 & 0xffffffffull);             // tensor_dim0_stride lo
    g1[6] = (int)((stride0 >> 32) & 0xffffull);         // stride hi (dim1_stride=0)
    g1[7] = 0;
    v4i z4 = {0, 0, 0, 0};
#if __clang_major__ >= 23
    v8i z8 = {0, 0, 0, 0, 0, 0, 0, 0};
    __builtin_amdgcn_tensor_load_to_lds(g0, g1, z4, z4, z8, 0);
#else
    __builtin_amdgcn_tensor_load_to_lds(g0, g1, z4, z4, 0);
#endif
}
__device__ __forceinline__ unsigned lds_off_of(const void* p) {
    return (unsigned)(unsigned long long)(size_t)p;     // flat addr low 32 = LDS offset
}
#endif

// ---------------------------------------------------------------------------
// f32 -> f16 conversion
// ---------------------------------------------------------------------------
__global__ void cvt_f32_f16_kernel(const float* __restrict__ src,
                                   _Float16* __restrict__ dst, int n) {
    int i = blockIdx.x * blockDim.x + threadIdx.x;
    if (i < n) dst[i] = (_Float16)src[i];
}

// ---------------------------------------------------------------------------
// Generic WMMA GEMM: C[M,N] = act( scale * (A[M,K] @ W[N,K]^T) + bias[N] )
// A, W row-major f16 with K contiguous. Block: 256 threads (8 waves),
// 128x128 output tile, K-step 32, double-buffered LDS tiles filled by the
// Tensor Data Mover (wave 0 posts descriptors; completion via TENSORcnt).
// Wave w: rows (w>>1)*32..+32, cols (w&1)*64..+64 -> 2x4 accum tiles.
// ---------------------------------------------------------------------------
template<bool OUT_F16, bool RELU>
__global__ __launch_bounds__(256)
void gemm_wmma_kernel(const _Float16* __restrict__ A,
                      const _Float16* __restrict__ W,
                      const float* __restrict__ bias,
                      void* __restrict__ Cout,
                      int M, int N, int K, float scale) {
    __shared__ _Float16 As[2][128][32];
    __shared__ _Float16 Ws[2][128][32];

    const int tid  = threadIdx.x;
    const int lane = tid & 31;
    const int wave = tid >> 5;
    const int h2   = lane >> 4;     // half-wave id
    const int ln   = lane & 15;
    const int wr   = wave >> 1;     // wave row 0..3
    const int wc   = wave & 1;      // wave col 0..1

    const long m0 = (long)blockIdx.x * 128;
    const long n0 = (long)blockIdx.y * 128;

    v8f acc[2][4] = {};

#if USE_TDM
    // prologue: post tile 0 into buffer 0
    if (wave == 0) {
        tdm_load_2d_f16(A + m0 * (long)K, lds_off_of(&As[0][0][0]),
                        (unsigned)K, (unsigned)M, 32, 128, (unsigned)K);
        tdm_load_2d_f16(W + n0 * (long)K, lds_off_of(&Ws[0][0][0]),
                        (unsigned)K, (unsigned)N, 32, 128, (unsigned)K);
    }
#else
    // cooperative loader mapping: 256 threads cover 128 rows x 32 halves
    const int lr = tid >> 1;            // 0..127
    const int lp = (tid & 1) * 16;      // 0 or 16 (halves)
#endif

    int it = 0;
    for (int k0 = 0; k0 < K; k0 += 32, ++it) {
        const int buf = it & 1;

#if USE_TDM
        if (wave == 0) {
            if (k0 + 32 < K) {
                // post next tile into the other buffer, then wait until only
                // those 2 ops remain outstanding (in-order => current tile done)
                tdm_load_2d_f16(A + m0 * (long)K + (k0 + 32),
                                lds_off_of(&As[buf ^ 1][0][0]),
                                (unsigned)K, (unsigned)M, 32, 128, (unsigned)K);
                tdm_load_2d_f16(W + n0 * (long)K + (k0 + 32),
                                lds_off_of(&Ws[buf ^ 1][0][0]),
                                (unsigned)K, (unsigned)N, 32, 128, (unsigned)K);
                __builtin_amdgcn_s_wait_tensorcnt(2);
            } else {
                __builtin_amdgcn_s_wait_tensorcnt(0);
            }
        }
        __syncthreads();    // current buffer visible to all waves
#else
        const v8h* ap = (const v8h*)(A + (m0 + lr) * (long)K + k0 + lp);
        v8h a0 = ap[0], a1 = ap[1];
        const v8h* wp = (const v8h*)(W + (n0 + lr) * (long)K + k0 + lp);
        v8h w0 = wp[0], w1 = wp[1];
        if (k0 + 32 < K) {
            __builtin_prefetch(A + (m0 + lr) * (long)K + k0 + 32 + lp, 0, 0);
            __builtin_prefetch(W + (n0 + lr) * (long)K + k0 + 32 + lp, 0, 0);
        }
        *(v8h*)&As[buf][lr][lp]     = a0;
        *(v8h*)&As[buf][lr][lp + 8] = a1;
        *(v8h*)&Ws[buf][lr][lp]     = w0;
        *(v8h*)&Ws[buf][lr][lp + 8] = w1;
        __syncthreads();
#endif

        // A fragments (2 row tiles): lane ln = M row, halves 0..7 -> K 16h2..,
        // halves 8..15 -> K 16h2+8..
        v16h af[2];
#pragma unroll
        for (int mi = 0; mi < 2; ++mi) {
            const int r = wr * 32 + mi * 16 + ln;
            v8h lo = *(const v8h*)&As[buf][r][16 * h2];
            v8h hi = *(const v8h*)&As[buf][r][16 * h2 + 8];
#pragma unroll
            for (int i = 0; i < 8; ++i) { af[mi][i] = lo[i]; af[mi][i + 8] = hi[i]; }
        }
        // B fragments (4 col tiles): lane = N col, contiguous 16 halves K=16h2..
        v16h wf[4];
#pragma unroll
        for (int ni = 0; ni < 4; ++ni) {
            const int n = wc * 64 + ni * 16 + ln;
            wf[ni] = *(const v16h*)&Ws[buf][n][16 * h2];
        }
#pragma unroll
        for (int mi = 0; mi < 2; ++mi)
#pragma unroll
            for (int ni = 0; ni < 4; ++ni)
                acc[mi][ni] = __builtin_amdgcn_wmma_f32_16x16x32_f16(
                    false, af[mi], false, wf[ni], (short)0, acc[mi][ni],
                    false, false);

        __syncthreads();    // all waves done reading buf before it is refilled
    }

    // epilogue: element e of acc -> row base + e + 8*h2, col tile + ln
#pragma unroll
    for (int mi = 0; mi < 2; ++mi) {
#pragma unroll
        for (int ni = 0; ni < 4; ++ni) {
            const long col = n0 + wc * 64 + ni * 16 + ln;
            const float bv = bias ? bias[col] : 0.0f;
            const long rbase = m0 + wr * 32 + mi * 16 + 8 * h2;
#pragma unroll
            for (int e = 0; e < 8; ++e) {
                float v = acc[mi][ni][e] * scale + bv;
                if (RELU) v = fmaxf(v, 0.0f);
                const long idx = (rbase + e) * (long)N + col;
                if (OUT_F16) ((_Float16*)Cout)[idx] = (_Float16)v;
                else         ((float*)Cout)[idx]    = v;
            }
        }
    }
}

// ---------------------------------------------------------------------------
// Flash attention, f16 WMMA. Q pre-scaled by 1/sqrt(hd).
// Block: (b*H+h, q_block of 128). 8 waves, wave = one 16-row q strip.
// K loop over 32-key chunks staged in LDS (V transposed).
// ---------------------------------------------------------------------------
__global__ __launch_bounds__(256)
void attn_wmma_kernel(const _Float16* __restrict__ Q,
                      const _Float16* __restrict__ Kb,
                      const _Float16* __restrict__ Vb,
                      const int* __restrict__ mask,
                      _Float16* __restrict__ Ctx) {
    const int T = 2048, E = 1024, HD = 64, H = 16;
    __shared__ _Float16 Ks[32][64];      // key chunk, row-major
    __shared__ _Float16 Vt[64][32];      // value chunk, transposed
    __shared__ _Float16 Ps[8][16][32];   // per-wave P tile

    const int bh = blockIdx.x;
    const int b  = bh / H;
    const int hh = bh % H;
    const int q0 = blockIdx.y * 128;
    const int tid  = threadIdx.x;
    const int wave = tid >> 5;
    const int lane = tid & 31;
    const int h2   = lane >> 4;
    const int ln   = lane & 15;

    const long base = (long)b * T * E + hh * HD;
    const int  qrow = q0 + wave * 16;

    // Q strip fragments (16 x 64 -> two K-steps of 32)
    v16h qa[2];
#pragma unroll
    for (int s = 0; s < 2; ++s) {
        const _Float16* qp = Q + base + (long)(qrow + ln) * E + s * 32;
        v8h lo = *(const v8h*)(qp + 16 * h2);
        v8h hi = *(const v8h*)(qp + 16 * h2 + 8);
#pragma unroll
        for (int i = 0; i < 8; ++i) { qa[s][i] = lo[i]; qa[s][i + 8] = hi[i]; }
    }

    v8f o[4] = {};
    float mrun[8], lrun[8];
#pragma unroll
    for (int e = 0; e < 8; ++e) { mrun[e] = -3e38f; lrun[e] = 0.0f; }

    const int lvr = tid >> 3;        // 0..31 : K/V chunk row
    const int lvc = (tid & 7) * 8;   // 0..56 : col chunk

    for (int kc = 0; kc < T; kc += 32) {
        v8h kv = *(const v8h*)(Kb + base + (long)(kc + lvr) * E + lvc);
        v8h vv = *(const v8h*)(Vb + base + (long)(kc + lvr) * E + lvc);
        __syncthreads();
        *(v8h*)&Ks[lvr][lvc] = kv;
#pragma unroll
        for (int i = 0; i < 8; ++i) Vt[lvc + i][lvr] = vv[i];
        __syncthreads();

        // scores: S(16x32) = Q(16x64) @ K_chunk^T, two 16-col tiles
        v8f sc[2] = {};
#pragma unroll
        for (int j = 0; j < 2; ++j) {
#pragma unroll
            for (int s = 0; s < 2; ++s) {
                v16h kf = *(const v16h*)&Ks[16 * j + ln][32 * s + 16 * h2];
                sc[j] = __builtin_amdgcn_wmma_f32_16x16x32_f16(
                    false, qa[s], false, kf, (short)0, sc[j], false, false);
            }
            const int km = mask[b * T + kc + 16 * j + ln];
            if (!km) {
#pragma unroll
                for (int e = 0; e < 8; ++e) sc[j][e] = -1e30f;
            }
        }

        // online softmax (rows live across 16 lanes of each half-wave)
#pragma unroll
        for (int e = 0; e < 8; ++e) {
            float rm = fmaxf(sc[0][e], sc[1][e]);
#pragma unroll
            for (int msk = 1; msk < 16; msk <<= 1)
                rm = fmaxf(rm, __shfl_xor(rm, msk, 32));
            const float mnew  = fmaxf(mrun[e], rm);
            const float alpha = __expf(mrun[e] - mnew);
            mrun[e] = mnew;
            const float p0 = __expf(sc[0][e] - mnew);
            const float p1 = __expf(sc[1][e] - mnew);
            float rs = p0 + p1;
#pragma unroll
            for (int msk = 1; msk < 16; msk <<= 1)
                rs += __shfl_xor(rs, msk, 32);
            lrun[e] = lrun[e] * alpha + rs;
#pragma unroll
            for (int nt = 0; nt < 4; ++nt) o[nt][e] *= alpha;
            Ps[wave][e + 8 * h2][ln]      = (_Float16)p0;
            Ps[wave][e + 8 * h2][ln + 16] = (_Float16)p1;
        }

        // P(16x32) as A fragment, V chunk (32x64) as B -> O += P@V
        v16h pa;
        {
            v8h lo = *(const v8h*)&Ps[wave][ln][16 * h2];
            v8h hi = *(const v8h*)&Ps[wave][ln][16 * h2 + 8];
#pragma unroll
            for (int i = 0; i < 8; ++i) { pa[i] = lo[i]; pa[i + 8] = hi[i]; }
        }
#pragma unroll
        for (int nt = 0; nt < 4; ++nt) {
            v16h vf = *(const v16h*)&Vt[16 * nt + ln][16 * h2];
            o[nt] = __builtin_amdgcn_wmma_f32_16x16x32_f16(
                false, pa, false, vf, (short)0, o[nt], false, false);
        }
    }

    // normalize, apply query-position mask, store ctx (f16)
#pragma unroll
    for (int e = 0; e < 8; ++e) {
        const int rg = qrow + e + 8 * h2;
        const int qm = mask[b * T + rg];
        const float inv = (qm && lrun[e] > 0.0f) ? 1.0f / lrun[e] : 0.0f;
#pragma unroll
        for (int nt = 0; nt < 4; ++nt)
            Ctx[base + (long)rg * E + 16 * nt + ln] = (_Float16)(o[nt][e] * inv);
    }
}

// ---------------------------------------------------------------------------
// LayerNorm (ddof=1, matches reference): y = (a+r - mean)/(sd+eps)*g + b
// One 256-thread block per row of 1024.
// ---------------------------------------------------------------------------
__global__ __launch_bounds__(256)
void ln_kernel(const float* __restrict__ A, const float* __restrict__ R,
               const float* __restrict__ g, const float* __restrict__ beta,
               float* __restrict__ Y, _Float16* __restrict__ Yh) {
    const int N = 1024;
    const long row = blockIdx.x;
    const float* a = A + row * N;
    const float* r = R + row * N;

    float x[4];
    float s = 0.0f, sq = 0.0f;
#pragma unroll
    for (int i = 0; i < 4; ++i) {
        const int c = threadIdx.x + i * 256;
        x[i] = a[c] + r[c];
        s  += x[i];
        sq += x[i] * x[i];
    }
#pragma unroll
    for (int m = 1; m < 32; m <<= 1) {
        s  += __shfl_xor(s,  m, 32);
        sq += __shfl_xor(sq, m, 32);
    }
    __shared__ float ss[8], ssq[8];
    const int wave = threadIdx.x >> 5, lane = threadIdx.x & 31;
    if (lane == 0) { ss[wave] = s; ssq[wave] = sq; }
    __syncthreads();
    if (wave == 0) {
        float ts = (lane < 8) ? ss[lane]  : 0.0f;
        float tq = (lane < 8) ? ssq[lane] : 0.0f;
#pragma unroll
        for (int m = 1; m < 8; m <<= 1) {
            ts += __shfl_xor(ts, m, 32);
            tq += __shfl_xor(tq, m, 32);
        }
        if (lane == 0) { ss[0] = ts; ssq[0] = tq; }
    }
    __syncthreads();
    const float mean = ss[0] / N;
    const float var  = (ssq[0] - N * mean * mean) / (N - 1);
    const float inv  = 1.0f / (sqrtf(fmaxf(var, 0.0f)) + 1e-5f);
#pragma unroll
    for (int i = 0; i < 4; ++i) {
        const int c = threadIdx.x + i * 256;
        const float v = (x[i] - mean) * inv * g[c] + beta[c];
        Y[row * N + c] = v;
        if (Yh) Yh[row * N + c] = (_Float16)v;
    }
}

// ---------------------------------------------------------------------------
// Orchestration
// ---------------------------------------------------------------------------
extern "C" void kernel_launch(void* const* d_in, const int* in_sizes, int n_in,
                              void* d_out, int out_size, void* d_ws, size_t ws_size,
                              hipStream_t stream) {
    (void)in_sizes; (void)n_in; (void)out_size; (void)ws_size;
    const int B = 4, T = 2048, E = 1024, F = 4096, H = 16;
    const int M = B * T;   // 8192

    const float* x   = (const float*)d_in[0];
    const int*   msk = (const int*)d_in[1];
    const float* Wq  = (const float*)d_in[2];
    const float* Wk  = (const float*)d_in[3];
    const float* Wv  = (const float*)d_in[4];
    const float* Wo  = (const float*)d_in[5];
    const float* w1  = (const float*)d_in[6];
    const float* b1  = (const float*)d_in[7];
    const float* w2  = (const float*)d_in[8];
    const float* b2  = (const float*)d_in[9];
    const float* lng = (const float*)d_in[10];
    const float* lnb = (const float*)d_in[11];
    float* out = (float*)d_out;

    char* ws = (char*)d_ws;
    const size_t MB = 1ull << 20;
    _Float16* xh   = (_Float16*)(ws + 0 * MB);     // 16 MB
    _Float16* wqh  = (_Float16*)(ws + 16 * MB);    //  2 MB
    _Float16* wkh  = (_Float16*)(ws + 18 * MB);
    _Float16* wvh  = (_Float16*)(ws + 20 * MB);
    _Float16* woh  = (_Float16*)(ws + 22 * MB);
    _Float16* w1h  = (_Float16*)(ws + 24 * MB);    //  8 MB
    _Float16* w2h  = (_Float16*)(ws + 32 * MB);    //  8 MB
    _Float16* Qh   = (_Float16*)(ws + 40 * MB);    // 16 MB
    _Float16* Kh   = (_Float16*)(ws + 56 * MB);    // 16 MB
    _Float16* Vh   = (_Float16*)(ws + 72 * MB);    // 16 MB
    _Float16* Ch   = (_Float16*)(ws + 88 * MB);    // 16 MB
    float*    attout = (float*)(ws + 104 * MB);    // 32 MB
    float*    hbuf   = (float*)(ws + 136 * MB);    // 32 MB
    _Float16* hh     = (_Float16*)(ws + 168 * MB); // 16 MB
    _Float16* ffh  = (_Float16*)(ws + 40 * MB);    // reuse Q/K/V/Ctx (64 MB)
    float*    ff2  = (float*)(ws + 104 * MB);      // reuse attout (32 MB)

    // --- f32 -> f16 conversions ---
    cvt_f32_f16_kernel<<<DIV_UP(M * E, 256), 256, 0, stream>>>(x, xh, M * E);
    cvt_f32_f16_kernel<<<DIV_UP(E * E, 256), 256, 0, stream>>>(Wq, wqh, E * E);
    cvt_f32_f16_kernel<<<DIV_UP(E * E, 256), 256, 0, stream>>>(Wk, wkh, E * E);
    cvt_f32_f16_kernel<<<DIV_UP(E * E, 256), 256, 0, stream>>>(Wv, wvh, E * E);
    cvt_f32_f16_kernel<<<DIV_UP(E * E, 256), 256, 0, stream>>>(Wo, woh, E * E);
    cvt_f32_f16_kernel<<<DIV_UP(F * E, 256), 256, 0, stream>>>(w1, w1h, F * E);
    cvt_f32_f16_kernel<<<DIV_UP(E * F, 256), 256, 0, stream>>>(w2, w2h, E * F);

    // --- Q/K/V projections (Q pre-scaled by 1/sqrt(64)=0.125) ---
    dim3 gEE(M / 128, E / 128);
    gemm_wmma_kernel<true, false><<<gEE, 256, 0, stream>>>(xh, wqh, nullptr, Qh, M, E, E, 0.125f);
    gemm_wmma_kernel<true, false><<<gEE, 256, 0, stream>>>(xh, wkh, nullptr, Kh, M, E, E, 1.0f);
    gemm_wmma_kernel<true, false><<<gEE, 256, 0, stream>>>(xh, wvh, nullptr, Vh, M, E, E, 1.0f);

    // --- flash attention ---
    attn_wmma_kernel<<<dim3(B * H, T / 128), 256, 0, stream>>>(Qh, Kh, Vh, msk, Ch);

    // --- output projection + LN1 (with residual x) ---
    gemm_wmma_kernel<false, false><<<gEE, 256, 0, stream>>>(Ch, woh, nullptr, attout, M, E, E, 1.0f);
    ln_kernel<<<M, 256, 0, stream>>>(attout, x, lng, lnb, hbuf, hh);

    // --- FFN ---
    gemm_wmma_kernel<true, true><<<dim3(M / 128, F / 128), 256, 0, stream>>>(hh, w1h, b1, ffh, M, F, E, 1.0f);
    gemm_wmma_kernel<false, false><<<gEE, 256, 0, stream>>>(ffh, w2h, b2, ff2, M, E, F, 1.0f);

    // --- LN2 (with residual h) -> output ---
    ln_kernel<<<M, 256, 0, stream>>>(ff2, hbuf, lng, lnb, out, nullptr);
}
